// PKEModel_55061480734778
// MI455X (gfx1250) — matrix operand order repaired and checked by reading, not compiled, not run on hardware
//
#include <hip/hip_runtime.h>
#include <math.h>

// ---------------------------------------------------------------------------
// GATv2 GNN forward for MI455X (gfx1250, wave32, WMMA 16x16x32 f16)
//
// Roofline: output store = 256MB -> ~11us HBM floor; node features (1MB f16)
// live in L2 so all edge gathers are L2 traffic; total GEMM work ~13.5 GFLOP
// -> trivially hidden behind memory with v_wmma_f32_16x16x32_f16.
// ---------------------------------------------------------------------------

typedef _Float16 f16;
typedef __attribute__((ext_vector_type(16))) _Float16 v16h;
typedef __attribute__((ext_vector_type(8)))  _Float16 v8h;
typedef __attribute__((ext_vector_type(8)))  float    v8f;
typedef __attribute__((ext_vector_type(4)))  float    v4f;

constexpr int kN  = 8192;
constexpr int kE  = 524288;
constexpr int kEP = kE + kN;     // edges + self loops for GAT layers
constexpr int kD  = 64;

// ----------------------------- helpers -------------------------------------

__device__ inline v8f vzero8() {
  v8f z;
#pragma unroll
  for (int i = 0; i < 8; ++i) z[i] = 0.f;
  return z;
}

__device__ inline v16h mk16(v8h lo, v8h hi) {
  v16h r;
#pragma unroll
  for (int i = 0; i < 8; ++i) { r[i] = lo[i]; r[i + 8] = hi[i]; }
  return r;
}

__device__ inline v8f wmma_f16(v16h a, v16h b, v8f c) {
  // D = A(16x32 f16) x B(32x16 f16) + C(16x16 f32)
  return __builtin_amdgcn_wmma_f32_16x16x32_f16(
      false, a, false, b, (short)0, c, false, false);
}

// order-preserving float<->uint for atomicMax-based segment max
__device__ inline unsigned f2ord(float f) {
  unsigned u = __float_as_uint(f);
  return (u & 0x80000000u) ? ~u : (u | 0x80000000u);
}
__device__ inline float ord2f(unsigned u) {
  unsigned v = (u & 0x80000000u) ? (u & 0x7fffffffu) : ~u;
  return __uint_as_float(v);
}

// Swizzle a row-major f32 weight [K][64] into WMMA B-fragment order:
// dst[frag*512 + lane*16 + i], frag = kk*4 + ntile, per-lane 16 f16 contiguous.
// B layout (16-bit 32x16): lane: n = l&15, h = l>>4; half i -> K = 8h + (i&7) + 16*(i>>3).
__device__ inline void swizzleB(const float* W, f16* dst, int nfrag,
                                int tid, int nth) {
  int total = nfrag * 512;
  for (int idx = tid; idx < total; idx += nth) {
    int f = idx >> 9, rem = idx & 511, lane = rem >> 4, i = rem & 15;
    int h = lane >> 4, n = lane & 15;
    int kk = f >> 2, t = f & 3;
    dst[idx] = (f16)W[(32 * kk + 8 * h + (i & 7) + 16 * (i >> 3)) * 64 + 16 * t + n];
  }
}

// ------------------------- 1: input linear 12->64 ---------------------------

__global__ void __launch_bounds__(256)
k_lin_in(const float* __restrict__ x, const float* __restrict__ W1,
         const float* __restrict__ b1, float* __restrict__ y) {
  int i = blockIdx.x * 256 + threadIdx.x;
  if (i >= kN * kD) return;
  int node = i >> 6, c = i & 63;
  float a = b1[c];
#pragma unroll
  for (int f = 0; f < 12; ++f) a += x[node * 12 + f] * W1[f * 64 + c];
  y[i] = a > 0.f ? a : 0.f;
}

// ------------- 2: WMMA [N,64]@[64,64] GEMM (plain / residual+relu) ----------
// MODE 0: Y = X@W + b        MODE 1: Y = X + relu(X@W + b)   (in-place OK)

template <int MODE>
__global__ void __launch_bounds__(256)
k_gemm64(const float* __restrict__ X, const float* __restrict__ W,
         const float* __restrict__ bias, float* __restrict__ Y) {
  __shared__ alignas(32) f16 ws[8 * 512];
  __shared__ float bs[64];
  int tid = threadIdx.x;
  swizzleB(W, ws, 8, tid, 256);
  if (tid < 64) bs[tid] = bias[tid];
  __syncthreads();

  int lane = tid & 31, wave = tid >> 5;
  int h = lane >> 4, n = lane & 15;
  int tile = blockIdx.x * 8 + wave;          // 64 blocks * 8 waves = 512 tiles
  if (tile >= kN / 16) return;
  int m0 = tile * 16;

  v8f acc[4];
#pragma unroll
  for (int t = 0; t < 4; ++t) acc[t] = vzero8();

#pragma unroll
  for (int kk = 0; kk < 2; ++kk) {
    // A frag: lane row = m0 + n; K runs [32kk+8h, +8) and [32kk+16+8h, +8)
    const float* rb = X + (size_t)(m0 + n) * 64 + 32 * kk + 8 * h;
    v4f f0 = *(const v4f*)(rb);
    v4f f1 = *(const v4f*)(rb + 4);
    v4f f2 = *(const v4f*)(rb + 16);
    v4f f3 = *(const v4f*)(rb + 20);
    v16h a;
#pragma unroll
    for (int i = 0; i < 4; ++i) {
      a[i] = (f16)f0[i]; a[i + 4] = (f16)f1[i];
      a[i + 8] = (f16)f2[i]; a[i + 12] = (f16)f3[i];
    }
#pragma unroll
    for (int t = 0; t < 4; ++t) {
      v16h b = *(const v16h*)(ws + ((kk * 4 + t) * 32 + lane) * 16);
      acc[t] = wmma_f16(a, b, acc[t]);
    }
  }

#pragma unroll
  for (int t = 0; t < 4; ++t) {
    float bv = bs[16 * t + n];
#pragma unroll
    for (int q = 0; q < 8; ++q) {
      int m = m0 + 8 * h + q;                // C layout: row = reg + 8h
      size_t o = (size_t)m * 64 + 16 * t + n;
      float g = acc[t][q] + bv;
      if (MODE == 0) {
        Y[o] = g;
      } else {
        float rl = g > 0.f ? g : 0.f;
        Y[o] = X[o] + rl;
      }
    }
  }
}

// ----------------------- 3: GATv2 per-edge kernels --------------------------

__global__ void __launch_bounds__(256)
k_edge_logits(const float* __restrict__ xl, const float* __restrict__ xr,
              const float* __restrict__ att, const int* __restrict__ src,
              const int* __restrict__ dst, float* __restrict__ logits,
              unsigned* __restrict__ segmax) {
  int e = blockIdx.x * 256 + threadIdx.x;
  if (e >= kEP) return;
  int s, d;
  if (e < kE) { s = src[e]; d = dst[e]; } else { s = d = e - kE; }
  const float* ps = xl + (size_t)s * 64;
  const float* pd = xr + (size_t)d * 64;
  float lg[4] = {0.f, 0.f, 0.f, 0.f};
#pragma unroll
  for (int c = 0; c < 64; ++c) {
    float v = ps[c] + pd[c];
    v = v > 0.f ? v : 0.2f * v;              // leaky_relu(0.2)
    lg[c >> 4] += v * att[c];                // att flat [h*16+c] == c
  }
#pragma unroll
  for (int h = 0; h < 4; ++h) {
    logits[(size_t)e * 4 + h] = lg[h];
    atomicMax(segmax + (size_t)d * 4 + h, f2ord(lg[h]));
  }
}

__global__ void __launch_bounds__(256)
k_edge_exp(const int* __restrict__ dst, float* __restrict__ logits,
           const unsigned* __restrict__ segmax, float* __restrict__ denom) {
  int e = blockIdx.x * 256 + threadIdx.x;
  if (e >= kEP) return;
  int d = (e < kE) ? dst[e] : (e - kE);
#pragma unroll
  for (int h = 0; h < 4; ++h) {
    float m = ord2f(segmax[(size_t)d * 4 + h]);
    float p = __expf(logits[(size_t)e * 4 + h] - m);
    logits[(size_t)e * 4 + h] = p;
    atomicAdd(denom + (size_t)d * 4 + h, p);
  }
}

__global__ void __launch_bounds__(256)
k_edge_scatter(const int* __restrict__ src, const int* __restrict__ dst,
               const float* __restrict__ xl, const float* __restrict__ p,
               const float* __restrict__ denom, float* __restrict__ acc) {
  int e = blockIdx.x * 256 + threadIdx.x;
  if (e >= kEP) return;
  int s, d;
  if (e < kE) { s = src[e]; d = dst[e]; } else { s = d = e - kE; }
#pragma unroll
  for (int h = 0; h < 4; ++h) {
    float alpha = p[(size_t)e * 4 + h] / denom[(size_t)d * 4 + h];
#pragma unroll
    for (int cl = 0; cl < 16; ++cl) {
      int c = h * 16 + cl;
      atomicAdd(acc + (size_t)d * 64 + c, alpha * xl[(size_t)s * 64 + c]);
    }
  }
}

__global__ void __launch_bounds__(256)
k_gat_epi(float* __restrict__ x, const float* __restrict__ acc,
          const float* __restrict__ bo) {
  int i = blockIdx.x * 256 + threadIdx.x;
  if (i < kN * kD) x[i] += acc[i] + bo[i & 63];
}

__global__ void __launch_bounds__(256)
k_cvt16(const float* __restrict__ x, f16* __restrict__ xf) {
  int i = blockIdx.x * 256 + threadIdx.x;
  if (i < kN * kD) xf[i] = (f16)x[i];
}

// ------------- 4: fused edge MLP (WMMA) + scatter into dense out ------------
// Per 16-edge tile (one wave):
//   ef1[16,64] = cat(x[s],x[d])[16,128] @ We1 + be1      (16 WMMA)
//   ef2        = relu(ef1 @ We2 + be2)                   (8 WMMA)
//   val        = sigmoid(ef2 @ We3 + be3)                (VALU + shfl reduce)
//   out[s*8192 + d] = val

__global__ void __launch_bounds__(256)
k_edge_mlp(const f16* __restrict__ xf, const float* __restrict__ We1,
           const float* __restrict__ be1, const float* __restrict__ We2,
           const float* __restrict__ be2, const float* __restrict__ We3,
           const float* __restrict__ be3, const int* __restrict__ src,
           const int* __restrict__ dst, float* __restrict__ out) {
  __shared__ alignas(32) f16 we1s[16 * 512];     // 16 B-frags (K=128, N=64)
  __shared__ alignas(32) f16 we2s[8 * 512];      //  8 B-frags (K=64,  N=64)
  __shared__ float we3f[64];
  __shared__ float be1s[64], be2s[64];
  __shared__ alignas(32) f16 stage[8 * 1024];    // per-wave 16x64 f16 tile

  int tid = threadIdx.x;
  swizzleB(We1, we1s, 16, tid, 256);
  swizzleB(We2, we2s, 8, tid, 256);
  if (tid < 64) { we3f[tid] = We3[tid]; be1s[tid] = be1[tid]; be2s[tid] = be2[tid]; }
  __syncthreads();

  const float bexp = be3[0];
  int lane = tid & 31, wave = tid >> 5;
  int h = lane >> 4, n = lane & 15;
  f16* st = stage + wave * 1024;

  int wglobal = blockIdx.x * 8 + wave;
  int nwaves = gridDim.x * 8;
  const int NT = kE / 16;                        // 32768 tiles

  for (int tile = wglobal; tile < NT; tile += nwaves) {
    int e = tile * 16 + n;                       // lane's A-frag row
    int s = src[e], d = dst[e];
    if (tile + nwaves < NT) {                    // gfx1250 global_prefetch_b8
      __builtin_prefetch(&src[(tile + nwaves) * 16], 0, 1);
      __builtin_prefetch(&dst[(tile + nwaves) * 16], 0, 1);
    }

    // ---- GEMM1: [16,128] @ We1[128,64] ----
    v8f acc[4];
#pragma unroll
    for (int t = 0; t < 4; ++t) acc[t] = vzero8();
#pragma unroll
    for (int kk = 0; kk < 4; ++kk) {             // kk 0,1 -> src; 2,3 -> dst
      const f16* base =
          xf + (size_t)((kk < 2) ? s : d) * 64 + 32 * (kk & 1) + 8 * h;
      v8h lo = *(const v8h*)(base);
      v8h hi = *(const v8h*)(base + 16);
      v16h a = mk16(lo, hi);
#pragma unroll
      for (int t = 0; t < 4; ++t) {
        v16h b = *(const v16h*)(we1s + ((kk * 4 + t) * 32 + lane) * 16);
        acc[t] = wmma_f16(a, b, acc[t]);
      }
    }

    // ---- bias, stage ef1 (C layout) into LDS as row-major [16][64] f16 ----
#pragma unroll
    for (int t = 0; t < 4; ++t) {
      float bv = be1s[16 * t + n];
#pragma unroll
      for (int q = 0; q < 8; ++q) {
        int m = 8 * h + q;
        st[m * 64 + 16 * t + n] = (f16)(acc[t][q] + bv);
      }
    }
    asm volatile("" ::: "memory");               // keep LDS store->load order

    // ---- GEMM2: ef1[16,64] @ We2[64,64] ----
    v8f acc2[4];
#pragma unroll
    for (int t = 0; t < 4; ++t) acc2[t] = vzero8();
#pragma unroll
    for (int kk = 0; kk < 2; ++kk) {
      const f16* base2 = st + n * 64 + 32 * kk + 8 * h;
      v8h lo = *(const v8h*)(base2);
      v8h hi = *(const v8h*)(base2 + 16);
      v16h a = mk16(lo, hi);
#pragma unroll
      for (int t = 0; t < 4; ++t) {
        v16h b = *(const v16h*)(we2s + ((kk * 4 + t) * 32 + lane) * 16);
        acc2[t] = wmma_f16(a, b, acc2[t]);
      }
    }

    // ---- relu + dot with We3 (partial per lane over its 4 n-columns) ----
    float part[8];
#pragma unroll
    for (int q = 0; q < 8; ++q) {
      float v = 0.f;
#pragma unroll
      for (int t = 0; t < 4; ++t) {
        float x2 = acc2[t][q] + be2s[16 * t + n];
        x2 = x2 > 0.f ? x2 : 0.f;
        v += x2 * we3f[16 * t + n];
      }
      part[q] = v;
    }
    // reduce across the 16 lanes of each half (masks < 16 stay in-half)
#pragma unroll
    for (int mask = 1; mask < 16; mask <<= 1) {
#pragma unroll
      for (int q = 0; q < 8; ++q)
        part[q] += __shfl_xor(part[q], mask, 32);
    }

    // lanes with n<8 store one edge each: row m = 8h + n
    if (n < 8) {
      int m = 8 * h + n;
      float val = part[0];
#pragma unroll
      for (int q = 1; q < 8; ++q) val = (n == q) ? part[q] : val;
      int e2 = tile * 16 + m;
      int s2 = src[e2], d2 = dst[e2];
      val = 1.f / (1.f + __expf(-(val + bexp)));
      out[(size_t)s2 * kN + d2] = val;
    }
  }
}

// ------------------------------ launcher ------------------------------------

extern "C" void kernel_launch(void* const* d_in, const int* in_sizes, int n_in,
                              void* d_out, int out_size, void* d_ws,
                              size_t ws_size, hipStream_t stream) {
  (void)in_sizes; (void)n_in; (void)out_size; (void)ws_size;

  const float* x   = (const float*)d_in[0];
  const float* W1  = (const float*)d_in[1];  const float* b1  = (const float*)d_in[2];
  const float* W2  = (const float*)d_in[3];  const float* b2  = (const float*)d_in[4];
  const float* Wl1 = (const float*)d_in[5];  const float* bl1 = (const float*)d_in[6];
  const float* Wr1 = (const float*)d_in[7];  const float* br1 = (const float*)d_in[8];
  const float* att1= (const float*)d_in[9];  const float* bo1 = (const float*)d_in[10];
  const float* W4  = (const float*)d_in[11]; const float* b4  = (const float*)d_in[12];
  const float* Wl2 = (const float*)d_in[13]; const float* bl2 = (const float*)d_in[14];
  const float* Wr2 = (const float*)d_in[15]; const float* br2 = (const float*)d_in[16];
  const float* att2= (const float*)d_in[17]; const float* bo2 = (const float*)d_in[18];
  const float* W5  = (const float*)d_in[19]; const float* b5  = (const float*)d_in[20];
  const float* We1 = (const float*)d_in[21]; const float* be1 = (const float*)d_in[22];
  const float* We2 = (const float*)d_in[23]; const float* be2 = (const float*)d_in[24];
  const float* We3 = (const float*)d_in[25]; const float* be3 = (const float*)d_in[26];
  const int*   src = (const int*)d_in[27];   const int*   dst = (const int*)d_in[28];

  // workspace carve-out (256B aligned)
  char* w = (char*)d_ws;
  size_t o = 0;
  auto carve = [&](size_t bytes) {
    size_t r = o; o += bytes; o = (o + 255) & ~(size_t)255; return r;
  };
  float*    xcur   = (float*)(w + carve((size_t)kN * 64 * 4));
  float*    xl     = (float*)(w + carve((size_t)kN * 64 * 4));
  float*    xr     = (float*)(w + carve((size_t)kN * 64 * 4));
  float*    gatacc = (float*)(w + carve((size_t)kN * 64 * 4));
  f16*      xf16   = (f16*)  (w + carve((size_t)kN * 64 * 2));
  float*    logits = (float*)(w + carve((size_t)kEP * 4 * 4));
  unsigned* segmax = (unsigned*)(w + carve((size_t)kN * 4 * 4));
  float*    denom  = (float*)(w + carve((size_t)kN * 4 * 4));

  const int gNode = (kN * kD) / 256;   // 2048
  const int gEdge = kEP / 256;         // 2080

  // zero the dense 8192x8192 output every call (harness poisons it)
  (void)hipMemsetAsync(d_out, 0, (size_t)kN * kN * 4, stream);

  // node encoder
  k_lin_in<<<gNode, 256, 0, stream>>>(x, W1, b1, xcur);
  k_gemm64<1><<<64, 256, 0, stream>>>(xcur, W2, b2, xcur);

  // ---- GATv2 layer 1 ----
  k_gemm64<0><<<64, 256, 0, stream>>>(xcur, Wl1, bl1, xl);
  k_gemm64<0><<<64, 256, 0, stream>>>(xcur, Wr1, br1, xr);
  (void)hipMemsetAsync(segmax, 0, (size_t)kN * 4 * 4, stream);
  (void)hipMemsetAsync(denom,  0, (size_t)kN * 4 * 4, stream);
  (void)hipMemsetAsync(gatacc, 0, (size_t)kN * 64 * 4, stream);
  k_edge_logits<<<gEdge, 256, 0, stream>>>(xl, xr, att1, src, dst, logits, segmax);
  k_edge_exp<<<gEdge, 256, 0, stream>>>(dst, logits, segmax, denom);
  k_edge_scatter<<<gEdge, 256, 0, stream>>>(src, dst, xl, logits, denom, gatacc);
  k_gat_epi<<<gNode, 256, 0, stream>>>(xcur, gatacc, bo1);

  k_gemm64<1><<<64, 256, 0, stream>>>(xcur, W4, b4, xcur);

  // ---- GATv2 layer 2 ----
  k_gemm64<0><<<64, 256, 0, stream>>>(xcur, Wl2, bl2, xl);
  k_gemm64<0><<<64, 256, 0, stream>>>(xcur, Wr2, br2, xr);
  (void)hipMemsetAsync(segmax, 0, (size_t)kN * 4 * 4, stream);
  (void)hipMemsetAsync(denom,  0, (size_t)kN * 4 * 4, stream);
  (void)hipMemsetAsync(gatacc, 0, (size_t)kN * 64 * 4, stream);
  k_edge_logits<<<gEdge, 256, 0, stream>>>(xl, xr, att2, src, dst, logits, segmax);
  k_edge_exp<<<gEdge, 256, 0, stream>>>(dst, logits, segmax, denom);
  k_edge_scatter<<<gEdge, 256, 0, stream>>>(src, dst, xl, logits, denom, gatacc);
  k_gat_epi<<<gNode, 256, 0, stream>>>(xcur, gatacc, bo2);

  k_gemm64<1><<<64, 256, 0, stream>>>(xcur, W5, b5, xcur);

  // ---- fused edge MLP -> dense output scatter ----
  k_cvt16<<<gNode, 256, 0, stream>>>(xcur, xf16);
  k_edge_mlp<<<512, 256, 0, stream>>>(xf16, We1, be1, We2, be2, We3, be3,
                                      src, dst, (float*)d_out);
}
